// TransformerBlock_9294309228601
// MI455X (gfx1250) — matrix operand order repaired and checked by reading, not compiled, not run on hardware
//
#include <hip/hip_runtime.h>
#include <math.h>

typedef __attribute__((ext_vector_type(16))) _Float16 v16h;
typedef __attribute__((ext_vector_type(8)))  _Float16 v8h;
typedef __attribute__((ext_vector_type(8)))  float    v8f;
typedef __attribute__((ext_vector_type(4)))  float    v4f;
typedef __attribute__((ext_vector_type(4)))  int      v4i;

#define BM 128
#define BN 128
#define BK 32
#define LDSW 40          // halfs per LDS row: 32 data + 8 pad (80B stride, spreads banks)
#define THREADS 128      // 4 waves of 32

#if defined(__has_builtin)
#if __has_builtin(__builtin_amdgcn_global_load_async_to_lds_b128) && \
    __has_builtin(__builtin_amdgcn_s_wait_asynccnt)
#define USE_ASYNC_LDS 1
#endif
#endif
#ifndef USE_ASYNC_LDS
#define USE_ASYNC_LDS 0
#endif

// ---------------------------------------------------------------- staging ---
struct Row32 { v8h q[4]; };   // one 32-wide K row held in registers as f16

__device__ __forceinline__ Row32 load_row(const float* p) {
  Row32 r;
#pragma unroll
  for (int i = 0; i < 4; ++i) {
    v4f a = ((const v4f*)p)[2 * i];
    v4f b = ((const v4f*)p)[2 * i + 1];
    v8h q;
    q[0] = (_Float16)a.x; q[1] = (_Float16)a.y; q[2] = (_Float16)a.z; q[3] = (_Float16)a.w;
    q[4] = (_Float16)b.x; q[5] = (_Float16)b.y; q[6] = (_Float16)b.z; q[7] = (_Float16)b.w;
    r.q[i] = q;
  }
  return r;
}
__device__ __forceinline__ Row32 load_row(const _Float16* p) {
  Row32 r;
#pragma unroll
  for (int i = 0; i < 4; ++i) r.q[i] = ((const v8h*)p)[i];
  return r;
}
__device__ __forceinline__ void store_row(_Float16* dst, const Row32& r) {
#pragma unroll
  for (int i = 0; i < 4; ++i) ((v8h*)dst)[i] = r.q[i];
}

#if USE_ASYNC_LDS
typedef __attribute__((address_space(1))) v4i gas_v4i;   // global (as1) int4
typedef __attribute__((address_space(3))) v4i las_v4i;   // LDS    (as3) int4

// DMA one 32-half row (64B) global -> LDS as 4 x b128 async transfers
__device__ __forceinline__ void async_row(const _Float16* g, _Float16* l) {
#pragma unroll
  for (int i = 0; i < 4; ++i)
    __builtin_amdgcn_global_load_async_to_lds_b128(
        (gas_v4i*)(g + i * 8), (las_v4i*)(l + i * 8), 0, 0);
}
#endif

// ------------------------------------------------ fragment load + compute ---
// A fragment per 16x32 ISA layout: lane half kh holds K={kh*8..+7, 16+kh*8..+7}
// B fragment: lane half kh holds K={kh*8..+7, 16+kh*8..+7} of column l16
// C/D: VGPR r holds M=r (lanes 0-15) / r+8 (lanes 16-31), N=l16
__device__ __forceinline__ void wave_mma_step(const _Float16* As, const _Float16* Bs,
                                              int wm, int wn, int l16, int kh,
                                              v8f (&acc)[4][4]) {
  v16h af[4], bf[4];
#pragma unroll
  for (int m = 0; m < 4; ++m) {
    const _Float16* rp = As + (wm + m * 16 + l16) * LDSW;
    ((v8h*)&af[m])[0] = *(const v8h*)(rp + kh * 8);
    ((v8h*)&af[m])[1] = *(const v8h*)(rp + 16 + kh * 8);
  }
#pragma unroll
  for (int n = 0; n < 4; ++n) {
    const _Float16* cp = Bs + (wn + n * 16 + l16) * LDSW;
    ((v8h*)&bf[n])[0] = *(const v8h*)(cp + kh * 8);
    ((v8h*)&bf[n])[1] = *(const v8h*)(cp + 16 + kh * 8);
  }
#pragma unroll
  for (int m = 0; m < 4; ++m)
#pragma unroll
    for (int n = 0; n < 4; ++n)
      acc[m][n] = __builtin_amdgcn_wmma_f32_16x16x32_f16(
          false, af[m], false, bf[n], (short)0, acc[m][n], false, false);
}

template <typename TOUT, bool HAS_BIAS>
__device__ __forceinline__ void wave_epilogue(const v8f (&acc)[4][4],
                                              const float* bias, TOUT* C,
                                              long rowBase, int colBase, int N,
                                              int wm, int wn, int l16, int kh,
                                              float scale) {
#pragma unroll
  for (int n = 0; n < 4; ++n) {
    const int col = colBase + wn + n * 16 + l16;
    const float bv = HAS_BIAS ? bias[col] : 0.0f;
#pragma unroll
    for (int m = 0; m < 4; ++m) {
      const long row0 = rowBase + wm + m * 16 + kh * 8;
#pragma unroll
      for (int r = 0; r < 8; ++r) {
        float v = acc[m][n][r] * scale + bv;
        C[(row0 + r) * (long)N + col] = (TOUT)v;
      }
    }
  }
}

// ------------------------------------------------------------- WMMA GEMM ---
// C[M,N] = scale * (A[M,K] @ B[N,K]^T) + bias[N]   (torch Linear convention)
// Register-staged variant (handles f32 inputs with convert-to-f16 on the fly).
template <typename TIN, typename TOUT, bool HAS_BIAS>
__global__ __launch_bounds__(THREADS)
void gemm_wmma(const TIN* __restrict__ A, const TIN* __restrict__ B,
               const float* __restrict__ bias, TOUT* __restrict__ C,
               int M, int N, int K, float scale) {
  __shared__ _Float16 smA[2][BM * LDSW];
  __shared__ _Float16 smB[2][BN * LDSW];

  const int tid  = threadIdx.x;
  const int lane = tid & 31;
  const int wave = tid >> 5;
  const int l16  = lane & 15;
  const int kh   = lane >> 4;
  const int wm = (wave & 1) * 64;
  const int wn = (wave >> 1) * 64;

  const long rowA = (long)blockIdx.y * BM + tid;
  const long rowB = (long)blockIdx.x * BN + tid;
  const TIN* pA = A + rowA * (long)K;
  const TIN* pB = B + rowB * (long)K;
  const int nk = K / BK;

  v8f acc[4][4];
#pragma unroll
  for (int m = 0; m < 4; ++m)
#pragma unroll
    for (int n = 0; n < 4; ++n) { v8f z = {}; acc[m][n] = z; }

  Row32 ra = load_row(pA);
  Row32 rb = load_row(pB);
  store_row(&smA[0][tid * LDSW], ra);
  store_row(&smB[0][tid * LDSW], rb);
  if (nk > 1) { ra = load_row(pA + BK); rb = load_row(pB + BK); }
  __syncthreads();

  int pb = 0;
  for (int kt = 0; kt < nk; ++kt) {
    wave_mma_step(smA[pb], smB[pb], wm, wn, l16, kh, acc);
    if (kt + 1 < nk) {
      const int np = pb ^ 1;
      store_row(&smA[np][tid * LDSW], ra);
      store_row(&smB[np][tid * LDSW], rb);
      if (kt + 2 < nk) {
        ra = load_row(pA + (long)(kt + 2) * BK);
        rb = load_row(pB + (long)(kt + 2) * BK);
      }
      if (kt + 3 < nk) {   // global_prefetch_b8 path
        __builtin_prefetch(pA + (long)(kt + 3) * BK, 0, 1);
        __builtin_prefetch(pB + (long)(kt + 3) * BK, 0, 1);
      }
      __syncthreads();
      pb = np;
    }
  }

  wave_epilogue<TOUT, HAS_BIAS>(acc, bias, C, (long)blockIdx.y * BM,
                                blockIdx.x * BN, N, wm, wn, l16, kh, scale);
}

#if USE_ASYNC_LDS
// Async-DMA variant for f16 inputs: global_load_async_to_lds_b128 feeds the
// double buffer directly, tracked with ASYNCcnt (no VGPR staging round-trip).
template <typename TOUT>
__global__ __launch_bounds__(THREADS)
void gemm_wmma_h_async(const _Float16* __restrict__ A, const _Float16* __restrict__ B,
                       TOUT* __restrict__ C, int M, int N, int K, float scale) {
  __shared__ _Float16 smA[2][BM * LDSW];
  __shared__ _Float16 smB[2][BN * LDSW];

  const int tid  = threadIdx.x;
  const int lane = tid & 31;
  const int wave = tid >> 5;
  const int l16  = lane & 15;
  const int kh   = lane >> 4;
  const int wm = (wave & 1) * 64;
  const int wn = (wave >> 1) * 64;

  const long rowA = (long)blockIdx.y * BM + tid;
  const long rowB = (long)blockIdx.x * BN + tid;
  const _Float16* pA = A + rowA * (long)K;
  const _Float16* pB = B + rowB * (long)K;
  const int nk = K / BK;

  v8f acc[4][4];
#pragma unroll
  for (int m = 0; m < 4; ++m)
#pragma unroll
    for (int n = 0; n < 4; ++n) { v8f z = {}; acc[m][n] = z; }

  // prologue: DMA tiles 0 and 1 into the two buffers (8 async ops each)
  async_row(pA, &smA[0][tid * LDSW]);
  async_row(pB, &smB[0][tid * LDSW]);
  if (nk > 1) {
    async_row(pA + BK, &smA[1][tid * LDSW]);
    async_row(pB + BK, &smB[1][tid * LDSW]);
    __builtin_amdgcn_s_wait_asynccnt(8);   // tile 0 complete (in-order retire)
  } else {
    __builtin_amdgcn_s_wait_asynccnt(0);
  }
  __syncthreads();

  int pb = 0;
  for (int kt = 0; kt < nk; ++kt) {
    wave_mma_step(smA[pb], smB[pb], wm, wn, l16, kh, acc);
    if (kt + 1 < nk) {
      __syncthreads();                     // all waves done reading smX[pb]
      if (kt + 2 < nk) {
        async_row(pA + (long)(kt + 2) * BK, &smA[pb][tid * LDSW]);
        async_row(pB + (long)(kt + 2) * BK, &smB[pb][tid * LDSW]);
        __builtin_amdgcn_s_wait_asynccnt(8);  // tile kt+1 done, kt+2 in flight
      } else {
        __builtin_amdgcn_s_wait_asynccnt(0);  // last refill already done
      }
      __syncthreads();                     // tile kt+1 visible to all waves
      pb ^= 1;
    }
  }

  wave_epilogue<TOUT, false>(acc, nullptr, C, (long)blockIdx.y * BM,
                             blockIdx.x * BN, N, wm, wn, l16, kh, scale);
}
#endif

// -------------------------------------------------------- f16 transpose ----
__global__ __launch_bounds__(256)
void transpose_h(const _Float16* __restrict__ in, _Float16* __restrict__ out,
                 int R, int Ccols) {
  __shared__ _Float16 t[32][33];
  const int bx = blockIdx.x * 32;
  const int by = blockIdx.y * 32;
  for (int i = threadIdx.y; i < 32; i += 8)
    t[i][threadIdx.x] = in[(long)(by + i) * Ccols + bx + threadIdx.x];
  __syncthreads();
  for (int i = threadIdx.y; i < 32; i += 8)
    out[(long)(bx + i) * R + by + threadIdx.x] = t[threadIdx.x][i];
}

// ---------------------------------------------------------- row softmax ----
__global__ __launch_bounds__(256)
void softmax_rows(_Float16* __restrict__ p, int S) {
  __shared__ float red[256];
  const int tid = threadIdx.x;
  _Float16* rp = p + (long)blockIdx.x * S;
  const int per = S / 256;

  float vals[16];
  float lmax = -3.0e38f;
  for (int i = 0; i < per; ++i) {
    vals[i] = (float)rp[tid + i * 256];
    lmax = fmaxf(lmax, vals[i]);
  }
  red[tid] = lmax;
  __syncthreads();
  for (int s = 128; s > 0; s >>= 1) {
    if (tid < s) red[tid] = fmaxf(red[tid], red[tid + s]);
    __syncthreads();
  }
  const float m = red[0];
  __syncthreads();

  float lsum = 0.0f;
  for (int i = 0; i < per; ++i) {
    vals[i] = __expf(vals[i] - m);
    lsum += vals[i];
  }
  red[tid] = lsum;
  __syncthreads();
  for (int s = 128; s > 0; s >>= 1) {
    if (tid < s) red[tid] += red[tid + s];
    __syncthreads();
  }
  const float inv = 1.0f / red[0];
  for (int i = 0; i < per; ++i)
    rp[tid + i * 256] = (_Float16)(vals[i] * inv);
}

// -------------------------------------------------------------- launcher ---
extern "C" void kernel_launch(void* const* d_in, const int* in_sizes, int n_in,
                              void* d_out, int out_size, void* d_ws, size_t ws_size,
                              hipStream_t stream) {
  const float* x  = (const float*)d_in[0];
  const float* wq = (const float*)d_in[1];
  const float* bq = (const float*)d_in[2];
  const float* wk = (const float*)d_in[3];
  const float* bk = (const float*)d_in[4];
  const float* wv = (const float*)d_in[5];
  const float* bv = (const float*)d_in[6];
  float* out = (float*)d_out;

  const int D = in_sizes[2];              // 1024
  const int S = in_sizes[0] / D;          // 4096
  const float inv_sqrt_d = 1.0f / sqrtf((float)D);

  _Float16* qh  = (_Float16*)d_ws;
  _Float16* khb = qh  + (size_t)S * D;
  _Float16* vh  = khb + (size_t)S * D;
  _Float16* vth = vh  + (size_t)S * D;
  _Float16* ph  = vth + (size_t)S * D;

  dim3 blk(THREADS);

  // 1) QKV projections (f32 in, f16 out) — register-staged variant
  dim3 gQKV(D / BN, S / BM);
  gemm_wmma<float, _Float16, true><<<gQKV, blk, 0, stream>>>(x, wq, bq, qh,  S, D, D, 1.0f);
  gemm_wmma<float, _Float16, true><<<gQKV, blk, 0, stream>>>(x, wk, bk, khb, S, D, D, 1.0f);
  gemm_wmma<float, _Float16, true><<<gQKV, blk, 0, stream>>>(x, wv, bv, vh,  S, D, D, 1.0f);

  // 2) v^T so P@V becomes the same "@B^T" GEMM shape
  transpose_h<<<dim3(D / 32, S / 32), dim3(32, 8), 0, stream>>>(vh, vth, S, D);

  // 3) scores = (q @ k^T) / sqrt(D)  -> f16 [S,S]
#if USE_ASYNC_LDS
  gemm_wmma_h_async<_Float16><<<dim3(S / BN, S / BM), blk, 0, stream>>>(
      qh, khb, ph, S, S, D, inv_sqrt_d);
#else
  gemm_wmma<_Float16, _Float16, false><<<dim3(S / BN, S / BM), blk, 0, stream>>>(
      qh, khb, nullptr, ph, S, S, D, inv_sqrt_d);
#endif

  // 4) softmax rows in place
  softmax_rows<<<S, 256, 0, stream>>>(ph, S);

  // 5) out = P @ v == P @ (v^T)^T  (f16 in, f32 out)
#if USE_ASYNC_LDS
  gemm_wmma_h_async<float><<<dim3(D / BN, S / BM), blk, 0, stream>>>(
      ph, vth, out, S, D, S, 1.0f);
#else
  gemm_wmma<_Float16, float, false><<<dim3(D / BN, S / BM), blk, 0, stream>>>(
      ph, vth, nullptr, out, S, D, S, 1.0f);
#endif
}
